// SwinTransformerBlock_63299228009320
// MI455X (gfx1250) — matrix operand order repaired and checked by reading, not compiled, not run on hardware
//
#include <hip/hip_runtime.h>
#include <hip/hip_bf16.h>
#include <math.h>

typedef __attribute__((ext_vector_type(16))) _Float16 v16h;
typedef __attribute__((ext_vector_type(8)))  _Float16 v8h;
typedef __attribute__((ext_vector_type(8)))  float    v8f;

// ---- problem constants ----
constexpr int kC  = 128;
constexpr int kH  = 56;
constexpr int kW  = 56;
constexpr int kWS = 7;
constexpr int kSS = 3;
constexpr int kNH = 4;
constexpr int kT  = 8;
constexpr int kB  = 4;
constexpr int kBT = kB * kT;           // 32
constexpr int kL  = kH * kW;           // 3136
constexpr int kN  = kWS * kWS;         // 49
constexpr int kHD = kC / kNH;          // 32
constexpr int kNW = (kH / kWS) * (kW / kWS); // 64
constexpr int kM  = kBT * kL;          // 100352 tokens
#define SCALE_F 0.17677669529663687f   // 32^-0.5

// =======================================================================
// device helpers
// =======================================================================
__device__ inline float dgelu(float x) {
    return 0.5f * x * (1.0f + erff(x * 0.70710678118654752f));
}

__device__ inline float wave_sum(float v) {
    #pragma unroll
    for (int off = 16; off > 0; off >>= 1) v += __shfl_xor(v, off, 32);
    return v;
}

__device__ inline int mask_region(int c) {   // canvas coord thresholds: H-WS=49, H-SS=53
    return c < 49 ? 0 : (c < 53 ? 1 : 2);
}

// LN over one row of 128 channels by one wave; write f16 row
__device__ inline void ln_row_f16(const float* xr, const float* g, const float* b,
                                  _Float16* outr) {
    const int lane = threadIdx.x & 31;
    float v[4];
    float s = 0.0f;
    #pragma unroll
    for (int i = 0; i < 4; ++i) { v[i] = xr[lane + 32 * i]; s += v[i]; }
    s = wave_sum(s);
    const float mean = s * (1.0f / 128.0f);
    float var = 0.0f;
    #pragma unroll
    for (int i = 0; i < 4; ++i) { float d = v[i] - mean; var += d * d; }
    var = wave_sum(var) * (1.0f / 128.0f);
    const float inv = rsqrtf(var + 1e-5f);
    #pragma unroll
    for (int i = 0; i < 4; ++i) {
        const int c = lane + 32 * i;
        outr[c] = (_Float16)((v[i] - mean) * inv * g[c] + b[c]);
    }
}

// =======================================================================
// weight transpose f32 (K x N) row-major -> f16 (N x K) row-major
// =======================================================================
__global__ void k_wt_f16(const float* __restrict__ in, _Float16* __restrict__ out,
                         int K, int N) {
    int idx = blockIdx.x * 256 + threadIdx.x;
    if (idx >= K * N) return;
    int n = idx % N, k = idx / N;
    out[(size_t)n * K + k] = (_Float16)in[(size_t)k * N + n];
}

// =======================================================================
// layernorm kernels (one wave per token row)
// =======================================================================
__global__ void k_ln_permute_t(const float* __restrict__ x, const float* __restrict__ g,
                               const float* __restrict__ b, _Float16* __restrict__ out) {
    const int m = blockIdx.x * 8 + (threadIdx.x >> 5);   // out row (b*L+l)*T + t
    const int t  = m & 7;
    const int bl = m >> 3;
    const int l  = bl % kL;
    const int bb = bl / kL;
    const float* xr = x + (size_t)((bb * kT + t) * kL + l) * kC;
    ln_row_f16(xr, g, b, out + (size_t)m * kC);
}

__global__ void k_ln_window(const float* __restrict__ x, const float* __restrict__ g,
                            const float* __restrict__ b, _Float16* __restrict__ out) {
    const int m = blockIdx.x * 8 + (threadIdx.x >> 5);   // row (bt*NW+w)*49 + n
    const int n   = m % kN;
    const int wdw = (m / kN) % kNW;
    const int bt  = m / (kN * kNW);
    const int i = n / kWS, j = n % kWS;
    const int wh = wdw >> 3, wwi = wdw & 7;
    const int hh = wh * kWS + i;                 // canvas (rolled) coords
    const int ww = wwi * kWS + j;
    const int h = (hh + kSS) % kH;               // source coords (roll by -SS)
    const int w = (ww + kSS) % kW;
    const float* xr = x + (size_t)(bt * kL + h * kW + w) * kC;
    ln_row_f16(xr, g, b, out + (size_t)m * kC);
}

__global__ void k_ln_plain(const float* __restrict__ x, const float* __restrict__ g,
                           const float* __restrict__ b, _Float16* __restrict__ out) {
    const int m = blockIdx.x * 8 + (threadIdx.x >> 5);
    ln_row_f16(x + (size_t)m * kC, g, b, out + (size_t)m * kC);
}

// =======================================================================
// WMMA GEMM: Cf32/Ch16 = act(A[f16, MxK] * WtT[f16, NxK stored N-major] + bias)
// 256 threads = 8 waves; block tile 32(M) x 64(N); wave tile 16x16.
// =======================================================================
__global__ void __launch_bounds__(256)
k_gemm(const _Float16* __restrict__ A, const _Float16* __restrict__ Wt,
       const float* __restrict__ bias, float* __restrict__ Cf,
       _Float16* __restrict__ Ch, int M, int N, int K, int act) {
    const int wave = threadIdx.x >> 5;
    const int lane = threadIdx.x & 31;
    const int row0 = blockIdx.x * 32 + (wave >> 2) * 16;
    const int col0 = blockIdx.y * 64 + (wave & 3) * 16;
    if (row0 >= M || col0 >= N) return;

    const int r  = lane & 15;
    const int hi = lane >> 4;
    // A layout (16x32 f16): lane<16 -> row=lane, K {0..7,16..23}; lane>=16 -> K {8..15,24..31}
    const _Float16* ap = A  + (size_t)(row0 + r) * K + hi * 8;
    // B layout (32x16 f16): lane<16 -> col=lane, K {0..7,16..23}; lane>=16 -> K {8..15,24..31}
    const _Float16* bp = Wt + (size_t)(col0 + r) * K + hi * 8;

    v8f acc;
    #pragma unroll
    for (int e = 0; e < 8; ++e) acc[e] = 0.0f;

    for (int k0 = 0; k0 < K; k0 += 32) {
        v8h a0 = *(const v8h*)(ap + k0);
        v8h a1 = *(const v8h*)(ap + k0 + 16);
        v8h b0 = *(const v8h*)(bp + k0);
        v8h b1 = *(const v8h*)(bp + k0 + 16);
        v16h av, bv;
        #pragma unroll
        for (int e = 0; e < 8; ++e) {
            av[e] = a0[e]; av[e + 8] = a1[e];
            bv[e] = b0[e]; bv[e + 8] = b1[e];
        }
        acc = __builtin_amdgcn_wmma_f32_16x16x32_f16(false, av, false, bv,
                                                     (short)0, acc, false, false);
    }

    // C/D layout: lane<16 -> N=lane, M=g ; lane>=16 -> N=lane-16, M=8+g
    const int n  = col0 + r;
    const int rb = row0 + hi * 8;
    const float bvl = bias ? bias[n] : 0.0f;
    #pragma unroll
    for (int g2 = 0; g2 < 8; ++g2) {
        float v = acc[g2] + bvl;
        if (act) v = dgelu(v);
        const size_t idx = (size_t)(rb + g2) * N + n;
        if (Cf) Cf[idx] = v;
        if (Ch) Ch[idx] = (_Float16)v;
    }
}

// =======================================================================
// temporal attention: per (b*L, head, t) thread; T=8 keys, online softmax
// qkv rows in (b*L+l)*T+t order, row = [q(128) k(128) v(128)] f16
// =======================================================================
__global__ void k_attn_temporal(const _Float16* __restrict__ qkv,
                                const float* __restrict__ tpb,
                                _Float16* __restrict__ out) {
    const int tid = blockIdx.x * 256 + threadIdx.x;  // kB*kL*kNH*kT threads
    const int t  = tid & 7;
    const int h  = (tid >> 3) & 3;
    const int bl = tid >> 5;
    const _Float16* base = qkv + (size_t)bl * kT * 384;

    float q[32];
    #pragma unroll
    for (int d = 0; d < 32; ++d)
        q[d] = (float)base[(size_t)t * 384 + h * 32 + d] * SCALE_F;

    float mx = -1e30f, ssum = 0.0f, o[32];
    #pragma unroll
    for (int d = 0; d < 32; ++d) o[d] = 0.0f;

    for (int t2 = 0; t2 < kT; ++t2) {
        const _Float16* kr = base + (size_t)t2 * 384 + 128 + h * 32;
        float s = tpb[(t - t2 + kT - 1) * kNH + h];
        #pragma unroll
        for (int d = 0; d < 32; ++d) s += q[d] * (float)kr[d];
        const float m2   = fmaxf(mx, s);
        const float corr = expf(mx - m2);
        const float wgt  = expf(s - m2);
        ssum = ssum * corr + wgt;
        const _Float16* vr = kr + 128;
        #pragma unroll
        for (int d = 0; d < 32; ++d) o[d] = o[d] * corr + wgt * (float)vr[d];
        mx = m2;
    }
    const float inv = 1.0f / ssum;
    _Float16* orow = out + (size_t)(bl * kT + t) * kC + h * 32;
    #pragma unroll
    for (int d = 0; d < 32; ++d) orow[d] = (_Float16)(o[d] * inv);
}

// =======================================================================
// spatial window attention: block per (bt, window, head), thread per query
// bias from RPI formula, shifted-window mask from canvas region IDs
// =======================================================================
__global__ void k_attn_spatial(const _Float16* __restrict__ qkv,
                               const float* __restrict__ rpb,
                               _Float16* __restrict__ out) {
    const int unit = blockIdx.x;              // bt*kNW*kNH units
    const int h   = unit & 3;
    const int wdw = (unit >> 2) & 63;
    const int bt  = unit >> 8;
    const int qi  = threadIdx.x;
    if (qi >= kN) return;

    const int m0 = (bt * kNW + wdw) * kN;
    const int i1 = qi / kWS, j1 = qi % kWS;
    const int wh = wdw >> 3, wwi = wdw & 7;
    const int rid1 = mask_region(wh * kWS + i1) * 3 + mask_region(wwi * kWS + j1);

    const _Float16* qr = qkv + (size_t)(m0 + qi) * 384 + h * 32;
    float q[32];
    #pragma unroll
    for (int d = 0; d < 32; ++d) q[d] = (float)qr[d] * SCALE_F;

    float mx = -1e30f, ssum = 0.0f, o[32];
    #pragma unroll
    for (int d = 0; d < 32; ++d) o[d] = 0.0f;

    for (int p = 0; p < kN; ++p) {
        const int i2 = p / kWS, j2 = p % kWS;
        const _Float16* kr = qkv + (size_t)(m0 + p) * 384 + 128 + h * 32;
        float s = rpb[((i1 - i2 + kWS - 1) * (2 * kWS - 1) + (j1 - j2 + kWS - 1)) * kNH + h];
        const int rid2 = mask_region(wh * kWS + i2) * 3 + mask_region(wwi * kWS + j2);
        if (rid2 != rid1) s -= 100.0f;
        #pragma unroll
        for (int d = 0; d < 32; ++d) s += q[d] * (float)kr[d];
        const float m2   = fmaxf(mx, s);
        const float corr = expf(mx - m2);
        const float wgt  = expf(s - m2);
        ssum = ssum * corr + wgt;
        const _Float16* vr = kr + 128;
        #pragma unroll
        for (int d = 0; d < 32; ++d) o[d] = o[d] * corr + wgt * (float)vr[d];
        mx = m2;
    }
    const float inv = 1.0f / ssum;
    _Float16* orow = out + (size_t)(m0 + qi) * kC + h * 32;
    #pragma unroll
    for (int d = 0; d < 32; ++d) orow[d] = (_Float16)(o[d] * inv);
}

// =======================================================================
// residual / permute / unwindow adds
// =======================================================================
__global__ void k_add_temporal(const float* __restrict__ x, const float* __restrict__ a2,
                               float* __restrict__ x1) {
    const int e = blockIdx.x * 256 + threadIdx.x;   // kM*kC elements
    const int c = e & 127;
    const int m = e >> 7;                           // token (b*T+t)*L + l
    const int l   = m % kL;
    const int btt = m / kL;
    const int t   = btt % kT;
    const int bb  = btt / kT;
    const size_t ar = (size_t)((bb * kL + l) * kT + t) * kC + c;
    x1[e] = x[e] + a2[ar];
}

__global__ void k_window_reverse_add(const float* __restrict__ x1,
                                     const float* __restrict__ proj,
                                     const float* __restrict__ ada,
                                     float* __restrict__ x2) {
    const int e = blockIdx.x * 256 + threadIdx.x;   // kM*kC elements
    const int c = e & 127;
    const int m = e >> 7;                           // token bt*L + h*W + w
    const int l  = m % kL;
    const int bt = m / kL;
    const int h = l / kW, w = l % kW;
    const int hh = (h + kH - kSS) % kH;             // canvas coords (roll by +SS inverse)
    const int ww = (w + kW - kSS) % kW;
    const int wdw = (hh / kWS) * (kW / kWS) + (ww / kWS);
    const int n   = (hh % kWS) * kWS + (ww % kWS);
    const size_t wr = (size_t)((bt * kNW + wdw) * kN + n) * kC + c;
    x2[e] = x1[e] + proj[wr] + ada[wr];
}

__global__ void k_final_add(const float* __restrict__ x2, const float* __restrict__ mlp,
                            const float* __restrict__ ada, float* __restrict__ out) {
    const int e = blockIdx.x * 256 + threadIdx.x;
    out[e] = x2[e] + mlp[e] + 0.5f * ada[e];
}

// =======================================================================
// host launcher
// =======================================================================
static inline void launch_gemm(const _Float16* A, const _Float16* Wt, const float* bias,
                               float* Cf, _Float16* Ch, int M, int N, int K, int act,
                               hipStream_t s) {
    dim3 grid((M + 31) / 32, (N + 63) / 64);
    k_gemm<<<grid, 256, 0, s>>>(A, Wt, bias, Cf, Ch, M, N, K, act);
}

extern "C" void kernel_launch(void* const* d_in, const int* in_sizes, int n_in,
                              void* d_out, int out_size, void* d_ws, size_t ws_size,
                              hipStream_t stream) {
    const float* x      = (const float*)d_in[0];
    const float* qkv_w  = (const float*)d_in[1];
    const float* qkv_b  = (const float*)d_in[2];
    const float* proj_w = (const float*)d_in[3];
    const float* proj_b = (const float*)d_in[4];
    const float* rpb    = (const float*)d_in[5];
    const float* tpb    = (const float*)d_in[6];
    const float* n1g    = (const float*)d_in[7];
    const float* n1b    = (const float*)d_in[8];
    const float* n2g    = (const float*)d_in[9];
    const float* n2b    = (const float*)d_in[10];
    const float* mlp_w1 = (const float*)d_in[11];
    const float* mlp_b1 = (const float*)d_in[12];
    const float* mlp_w2 = (const float*)d_in[13];
    const float* mlp_b2 = (const float*)d_in[14];
    const float* s_w1   = (const float*)d_in[15];
    const float* s_b1   = (const float*)d_in[16];
    const float* s_w2   = (const float*)d_in[17];
    const float* s_b2   = (const float*)d_in[18];
    const float* t_w1   = (const float*)d_in[19];
    const float* t_b1   = (const float*)d_in[20];
    const float* t_w2   = (const float*)d_in[21];
    const float* t_b2   = (const float*)d_in[22];
    const float* a_w1   = (const float*)d_in[23];
    const float* a_b1   = (const float*)d_in[24];
    const float* a_w2   = (const float*)d_in[25];
    const float* a_b2   = (const float*)d_in[26];
    float* out = (float*)d_out;
    (void)in_sizes; (void)n_in; (void)out_size; (void)ws_size;

    // ---- carve workspace ----
    size_t off = 0;
    auto carve = [&](size_t bytes) -> void* {
        void* p = (char*)d_ws + off;
        off += (bytes + 255) & ~(size_t)255;
        return p;
    };
    const int Ah = 32;
    _Float16* qkv_wt = (_Float16*)carve((size_t)384 * 128 * 2);
    _Float16* proj_wt= (_Float16*)carve((size_t)128 * 128 * 2);
    _Float16* mlp1_wt= (_Float16*)carve((size_t)512 * 128 * 2);
    _Float16* mlp2_wt= (_Float16*)carve((size_t)128 * 512 * 2);
    _Float16* t1_wt  = (_Float16*)carve((size_t)Ah * 128 * 2);
    _Float16* t2_wt  = (_Float16*)carve((size_t)128 * Ah * 2);
    _Float16* s1_wt  = (_Float16*)carve((size_t)Ah * 128 * 2);
    _Float16* s2_wt  = (_Float16*)carve((size_t)128 * Ah * 2);
    _Float16* a1_wt  = (_Float16*)carve((size_t)Ah * 128 * 2);
    _Float16* a2_wt  = (_Float16*)carve((size_t)128 * Ah * 2);
    _Float16* H1 = (_Float16*)carve((size_t)kM * 128 * 2);   // LN outputs (f16)
    _Float16* HQ = (_Float16*)carve((size_t)kM * 512 * 2);   // qkv(384) / mlp hidden(512)
    _Float16* H3 = (_Float16*)carve((size_t)kM * 128 * 2);   // attention output (f16)
    _Float16* H4 = (_Float16*)carve((size_t)kM * 128 * 2);   // proj output (f16 mirror)
    _Float16* H5 = (_Float16*)carve((size_t)kM * 32 * 2);    // adapter hidden (f16)
    float* F1 = (float*)carve((size_t)kM * 128 * 4);         // proj_s / mlp_out
    float* F2 = (float*)carve((size_t)kM * 128 * 4);         // adapter outputs
    float* F3 = (float*)carve((size_t)kM * 128 * 4);         // x1
    float* F4 = (float*)carve((size_t)kM * 128 * 4);         // x2

    auto tr = [&](const float* in, _Float16* o, int K, int N) {
        k_wt_f16<<<(K * N + 255) / 256, 256, 0, stream>>>(in, o, K, N);
    };
    tr(qkv_w, qkv_wt, 128, 384);  tr(proj_w, proj_wt, 128, 128);
    tr(mlp_w1, mlp1_wt, 128, 512); tr(mlp_w2, mlp2_wt, 512, 128);
    tr(t_w1, t1_wt, 128, Ah);     tr(t_w2, t2_wt, Ah, 128);
    tr(s_w1, s1_wt, 128, Ah);     tr(s_w2, s2_wt, Ah, 128);
    tr(a_w1, a1_wt, 128, Ah);     tr(a_w2, a2_wt, Ah, 128);

    const int lnBlocks  = kM / 8;          // 12544 (8 waves per block)
    const int eltBlocks = kM * 128 / 256;  // 50176

    // ---- temporal attention branch ----
    k_ln_permute_t<<<lnBlocks, 256, 0, stream>>>(x, n1g, n1b, H1);
    launch_gemm(H1, qkv_wt, qkv_b, nullptr, HQ, kM, 384, 128, 0, stream);
    k_attn_temporal<<<kB * kL * kNH * kT / 256, 256, 0, stream>>>(HQ, tpb, H3);
    launch_gemm(H3, proj_wt, proj_b, nullptr, H4, kM, 128, 128, 0, stream);
    launch_gemm(H4, t1_wt, t_b1, nullptr, H5, kM, Ah, 128, 1, stream);   // GELU
    launch_gemm(H5, t2_wt, t_b2, F2, nullptr, kM, 128, Ah, 0, stream);
    k_add_temporal<<<eltBlocks, 256, 0, stream>>>(x, F2, F3);

    // ---- shifted-window spatial attention branch ----
    k_ln_window<<<lnBlocks, 256, 0, stream>>>(F3, n1g, n1b, H1);
    launch_gemm(H1, qkv_wt, qkv_b, nullptr, HQ, kM, 384, 128, 0, stream);
    k_attn_spatial<<<kBT * kNW * kNH, 64, 0, stream>>>(HQ, rpb, H3);
    launch_gemm(H3, proj_wt, proj_b, F1, H4, kM, 128, 128, 0, stream);
    launch_gemm(H4, s1_wt, s_b1, nullptr, H5, kM, Ah, 128, 1, stream);   // GELU
    launch_gemm(H5, s2_wt, s_b2, F2, nullptr, kM, 128, Ah, 0, stream);
    k_window_reverse_add<<<eltBlocks, 256, 0, stream>>>(F3, F1, F2, F4);

    // ---- MLP + MLP adapter ----
    k_ln_plain<<<lnBlocks, 256, 0, stream>>>(F4, n2g, n2b, H1);
    launch_gemm(H1, mlp1_wt, mlp_b1, nullptr, HQ, kM, 512, 128, 1, stream); // GELU
    launch_gemm(HQ, mlp2_wt, mlp_b2, F1, nullptr, kM, 128, 512, 0, stream);
    launch_gemm(H1, a1_wt, a_b1, nullptr, H5, kM, Ah, 128, 1, stream);      // GELU
    launch_gemm(H5, a2_wt, a_b2, F2, nullptr, kM, 128, Ah, 0, stream);
    k_final_add<<<eltBlocks, 256, 0, stream>>>(F4, F1, F2, out);
}
